// nBRCModel_10660108828787
// MI455X (gfx1250) — compile-verified, matrix operand order checked by reading, not run on hardware
//
#include <hip/hip_runtime.h>
#include <math.h>

#define DIN 512
#define HID 512
#define DOUT 512
#define BB 64
#define TT 512
#define LL 2
#define BT (BB*TT)

typedef __attribute__((ext_vector_type(16))) __bf16 v16bf;
typedef __attribute__((ext_vector_type(8)))  __bf16 v8bf;
typedef __attribute__((ext_vector_type(8)))  float  v8f;

// ---------------- fp32 -> bf16 split (hi + lo) --------------------------------------
__global__ void split_bf16_kernel(const float* __restrict__ in,
                                  __bf16* __restrict__ hi,
                                  __bf16* __restrict__ lo, int n) {
  int i = blockIdx.x * blockDim.x + threadIdx.x;
  if (i < n) {
    float x = in[i];
    __bf16 h = (__bf16)x;
    float r = x - (float)h;
    hi[i] = h;
    lo[i] = (__bf16)r;
  }
}

__global__ void zero_state_kernel(float* __restrict__ h,
                                  __bf16* __restrict__ hhi,
                                  __bf16* __restrict__ hlo, int n) {
  int i = blockIdx.x * blockDim.x + threadIdx.x;
  if (i < n) { h[i] = 0.0f; hhi[i] = (__bf16)0.0f; hlo[i] = (__bf16)0.0f; }
}

// ---------------- WMMA fragment load (wave32 layout) --------------------------------
// A (16x32 bf16): lane m in [0,16) holds row m; kh=0 -> K {0..7,16..23}, kh=1 -> +8.
// B (32x16 bf16) is the mirror with lane = N column (rows of W, [N,K] row-major).
__device__ __forceinline__ v16bf load_frag(const __bf16* row, int kBase, int kh) {
  const __bf16* p0 = row + kBase + kh * 8;
  v8bf a = *reinterpret_cast<const v8bf*>(p0);
  v8bf b = *reinterpret_cast<const v8bf*>(p0 + 16);
  v16bf r;
#pragma unroll
  for (int i = 0; i < 8; ++i) { r[i] = a[i]; r[i + 8] = b[i]; }
  return r;
}

__device__ __forceinline__ v8f wmma_bf16(v16bf a, v16bf b, v8f c) {
  // (neg_a, A, neg_b, B, c_mod, C, reuse_a, reuse_b)
  return __builtin_amdgcn_wmma_f32_16x16x32_bf16(false, a, false, b, (short)0, c,
                                                 false, false);
}

// ---------------- triple NT-GEMM: O{0,1,2} = A * W{0,1,2}^T + b{0,1,2} --------------
// 512 threads = 16 waves; block stages one 16x512 A tile (hi+lo) in LDS, each wave
// owns one of 16 column tiles. Split-bf16: hh + hl + lh into fp32 accumulators.
__global__ __launch_bounds__(512)
void proj3_wmma_kernel(const __bf16* __restrict__ Ahi, const __bf16* __restrict__ Alo,
                       const __bf16* __restrict__ W0hi, const __bf16* __restrict__ W0lo,
                       const __bf16* __restrict__ W1hi, const __bf16* __restrict__ W1lo,
                       const __bf16* __restrict__ W2hi, const __bf16* __restrict__ W2lo,
                       const float* __restrict__ b0, const float* __restrict__ b1,
                       const float* __restrict__ b2,
                       float* __restrict__ O0, float* __restrict__ O1,
                       float* __restrict__ O2) {
  __shared__ __bf16 sAhi[16 * DIN];
  __shared__ __bf16 sAlo[16 * DIN];

  const int tid  = threadIdx.x;
  const int lane = tid & 31;
  const int wave = tid >> 5;                 // 0..15
  const int idx  = lane & 15;
  const int kh   = lane >> 4;
  const int rowTile = blockIdx.x;
  const int colTile = blockIdx.y * 16 + wave;

  // cooperative stage: 16 rows x 512 = 1024 x v8bf chunks per split, 2 per thread
  {
    const v8bf* gH = reinterpret_cast<const v8bf*>(Ahi + (size_t)rowTile * 16 * DIN);
    const v8bf* gL = reinterpret_cast<const v8bf*>(Alo + (size_t)rowTile * 16 * DIN);
    v8bf* sH = reinterpret_cast<v8bf*>(sAhi);
    v8bf* sL = reinterpret_cast<v8bf*>(sAlo);
    sH[tid]       = gH[tid];
    sH[tid + 512] = gH[tid + 512];
    sL[tid]       = gL[tid];
    sL[tid + 512] = gL[tid + 512];
  }
  __syncthreads();

  const __bf16* ah_row = sAhi + idx * DIN;
  const __bf16* al_row = sAlo + idx * DIN;
  const __bf16* w0h = W0hi + (size_t)(colTile * 16 + idx) * DIN;
  const __bf16* w0l = W0lo + (size_t)(colTile * 16 + idx) * DIN;
  const __bf16* w1h = W1hi + (size_t)(colTile * 16 + idx) * DIN;
  const __bf16* w1l = W1lo + (size_t)(colTile * 16 + idx) * DIN;
  const __bf16* w2h = W2hi + (size_t)(colTile * 16 + idx) * DIN;
  const __bf16* w2l = W2lo + (size_t)(colTile * 16 + idx) * DIN;

  v8f acc0 = {}, acc1 = {}, acc2 = {};
  for (int k = 0; k < DIN; k += 32) {
    if (k + 64 < DIN) {
      __builtin_prefetch(w0h + k + 64, 0, 1);
      __builtin_prefetch(w1h + k + 64, 0, 1);
      __builtin_prefetch(w2h + k + 64, 0, 1);
    }
    v16bf aH = load_frag(ah_row, k, kh);
    v16bf aL = load_frag(al_row, k, kh);

    v16bf bH = load_frag(w0h, k, kh);
    v16bf bL = load_frag(w0l, k, kh);
    acc0 = wmma_bf16(aH, bH, acc0);
    acc0 = wmma_bf16(aH, bL, acc0);
    acc0 = wmma_bf16(aL, bH, acc0);

    bH = load_frag(w1h, k, kh);
    bL = load_frag(w1l, k, kh);
    acc1 = wmma_bf16(aH, bH, acc1);
    acc1 = wmma_bf16(aH, bL, acc1);
    acc1 = wmma_bf16(aL, bH, acc1);

    bH = load_frag(w2h, k, kh);
    bL = load_frag(w2l, k, kh);
    acc2 = wmma_bf16(aH, bH, acc2);
    acc2 = wmma_bf16(aH, bL, acc2);
    acc2 = wmma_bf16(aL, bH, acc2);
  }

  const int n  = colTile * 16 + idx;
  const int m0 = rowTile * 16 + kh * 8;
  const float bb0 = b0[n], bb1 = b1[n], bb2 = b2[n];
#pragma unroll
  for (int j = 0; j < 8; ++j) {
    size_t o = (size_t)(m0 + j) * HID + n;
    O0[o] = acc0[j] + bb0;
    O1[o] = acc1[j] + bb1;
    O2[o] = acc2[j] + bb2;
  }
}

// ---------------- single NT-GEMM (final FC): O = A * W^T + b ------------------------
__global__ __launch_bounds__(512)
void proj1_wmma_kernel(const __bf16* __restrict__ Ahi, const __bf16* __restrict__ Alo,
                       const __bf16* __restrict__ Whi, const __bf16* __restrict__ Wlo,
                       const float* __restrict__ bias, float* __restrict__ O) {
  __shared__ __bf16 sAhi[16 * HID];
  __shared__ __bf16 sAlo[16 * HID];

  const int tid  = threadIdx.x;
  const int lane = tid & 31;
  const int wave = tid >> 5;
  const int idx  = lane & 15;
  const int kh   = lane >> 4;
  const int rowTile = blockIdx.x;
  const int colTile = blockIdx.y * 16 + wave;

  {
    const v8bf* gH = reinterpret_cast<const v8bf*>(Ahi + (size_t)rowTile * 16 * HID);
    const v8bf* gL = reinterpret_cast<const v8bf*>(Alo + (size_t)rowTile * 16 * HID);
    v8bf* sH = reinterpret_cast<v8bf*>(sAhi);
    v8bf* sL = reinterpret_cast<v8bf*>(sAlo);
    sH[tid]       = gH[tid];
    sH[tid + 512] = gH[tid + 512];
    sL[tid]       = gL[tid];
    sL[tid + 512] = gL[tid + 512];
  }
  __syncthreads();

  const __bf16* ah_row = sAhi + idx * HID;
  const __bf16* al_row = sAlo + idx * HID;
  const __bf16* wh_row = Whi + (size_t)(colTile * 16 + idx) * HID;
  const __bf16* wl_row = Wlo + (size_t)(colTile * 16 + idx) * HID;

  v8f acc = {};
  for (int k = 0; k < HID; k += 32) {
    if (k + 64 < HID) {
      __builtin_prefetch(wh_row + k + 64, 0, 1);
      __builtin_prefetch(wl_row + k + 64, 0, 1);
    }
    v16bf aH = load_frag(ah_row, k, kh);
    v16bf aL = load_frag(al_row, k, kh);
    v16bf bH = load_frag(wh_row, k, kh);
    v16bf bL = load_frag(wl_row, k, kh);
    acc = wmma_bf16(aH, bH, acc);
    acc = wmma_bf16(aH, bL, acc);
    acc = wmma_bf16(aL, bH, acc);
  }

  const int n  = colTile * 16 + idx;
  const int m0 = rowTile * 16 + kh * 8;
  const float bb = bias[n];
#pragma unroll
  for (int j = 0; j < 8; ++j) {
    O[(size_t)(m0 + j) * DOUT + n] = acc[j] + bb;
  }
}

// ---------------- fused recurrent step: both gate GEMMs + nBRC update ---------------
// 8 blocks x 16 waves: rowTile = bid&3 (batch), colGroup = bid>>2. h-state tile is
// staged in LDS once per block; both gate GEMMs share the A fragments.
__global__ __launch_bounds__(512)
void nbrc_step_kernel(const float* __restrict__ pa, const float* __restrict__ pc,
                      const float* __restrict__ pu,
                      const __bf16* __restrict__ Wahi, const __bf16* __restrict__ Walo,
                      const __bf16* __restrict__ Wchi, const __bf16* __restrict__ Wclo,
                      const float* __restrict__ Wab, const float* __restrict__ Wcb,
                      const float* __restrict__ h_in,
                      const __bf16* __restrict__ hhi_in, const __bf16* __restrict__ hlo_in,
                      float* __restrict__ h_out,
                      __bf16* __restrict__ hhi_out, __bf16* __restrict__ hlo_out,
                      __bf16* __restrict__ out_hi, __bf16* __restrict__ out_lo,
                      int t) {
  __shared__ __bf16 sAhi[16 * HID];
  __shared__ __bf16 sAlo[16 * HID];

  const int tid  = threadIdx.x;
  const int lane = tid & 31;
  const int wave = tid >> 5;
  const int idx  = lane & 15;
  const int kh   = lane >> 4;
  const int rowTile = blockIdx.x & 3;                  // 64/16 batch tiles
  const int colTile = (blockIdx.x >> 2) * 16 + wave;   // 512/16 hidden tiles

  {
    const v8bf* gH = reinterpret_cast<const v8bf*>(hhi_in + (size_t)rowTile * 16 * HID);
    const v8bf* gL = reinterpret_cast<const v8bf*>(hlo_in + (size_t)rowTile * 16 * HID);
    v8bf* sH = reinterpret_cast<v8bf*>(sAhi);
    v8bf* sL = reinterpret_cast<v8bf*>(sAlo);
    sH[tid]       = gH[tid];
    sH[tid + 512] = gH[tid + 512];
    sL[tid]       = gL[tid];
    sL[tid + 512] = gL[tid + 512];
  }
  __syncthreads();

  const __bf16* ah_row = sAhi + idx * HID;
  const __bf16* al_row = sAlo + idx * HID;
  const __bf16* wah = Wahi + (size_t)(colTile * 16 + idx) * HID;
  const __bf16* wal = Walo + (size_t)(colTile * 16 + idx) * HID;
  const __bf16* wch = Wchi + (size_t)(colTile * 16 + idx) * HID;
  const __bf16* wcl = Wclo + (size_t)(colTile * 16 + idx) * HID;

  v8f accA = {}, accC = {};
  for (int k = 0; k < HID; k += 32) {
    v16bf aH = load_frag(ah_row, k, kh);
    v16bf aL = load_frag(al_row, k, kh);

    v16bf bH = load_frag(wah, k, kh);
    v16bf bL = load_frag(wal, k, kh);
    accA = wmma_bf16(aH, bH, accA);
    accA = wmma_bf16(aH, bL, accA);
    accA = wmma_bf16(aL, bH, accA);

    bH = load_frag(wch, k, kh);
    bL = load_frag(wcl, k, kh);
    accC = wmma_bf16(aH, bH, accC);
    accC = wmma_bf16(aH, bL, accC);
    accC = wmma_bf16(aL, bH, accC);
  }

  const int n  = colTile * 16 + idx;
  const int m0 = rowTile * 16 + kh * 8;
  const float wab = Wab[n], wcb = Wcb[n];
#pragma unroll
  for (int j = 0; j < 8; ++j) {
    const int m = m0 + j;
    const size_t hoff = (size_t)m * HID + n;
    const size_t poff = ((size_t)m * TT + t) * HID + n;
    const float hold = h_in[hoff];
    const float a = 1.0f + tanhf(pa[poff] + accA[j] + wab);
    const float c = 1.0f / (1.0f + expf(-(pc[poff] + accC[j] + wcb)));
    const float u = tanhf(pu[poff] + a * hold);
    const float hn = c * hold + (1.0f - c) * u;
    h_out[hoff] = hn;
    __bf16 hb = (__bf16)hn;
    __bf16 lb = (__bf16)(hn - (float)hb);
    hhi_out[hoff] = hb;
    hlo_out[hoff] = lb;
    out_hi[poff] = hb;
    out_lo[poff] = lb;
  }
}

// ------------------------------------------------------------------------------------
extern "C" void kernel_launch(void* const* d_in, const int* in_sizes, int n_in,
                              void* d_out, int out_size, void* d_ws, size_t ws_size,
                              hipStream_t stream) {
  const float* x    = (const float*)d_in[0];
  const float* Ua_w = (const float*)d_in[1];
  const float* Ua_b = (const float*)d_in[2];
  const float* Wa_w = (const float*)d_in[3];
  const float* Wa_b = (const float*)d_in[4];
  const float* Uc_w = (const float*)d_in[5];
  const float* Uc_b = (const float*)d_in[6];
  const float* Wc_w = (const float*)d_in[7];
  const float* Wc_b = (const float*)d_in[8];
  const float* U_w  = (const float*)d_in[9];
  const float* U_b  = (const float*)d_in[10];
  const float* fc_w = (const float*)d_in[11];
  const float* fc_b = (const float*)d_in[12];

  char* wsp = (char*)d_ws;
  size_t off = 0;
  auto carve = [&](size_t bytes) -> char* {
    char* p = wsp + off;
    off += (bytes + 255) & ~(size_t)255;
    return p;
  };

  const size_t WMAT = (size_t)LL * HID * DIN;  // both layers stacked
  __bf16* Ua_hi = (__bf16*)carve(WMAT * 2);
  __bf16* Ua_lo = (__bf16*)carve(WMAT * 2);
  __bf16* Wa_hi = (__bf16*)carve(WMAT * 2);
  __bf16* Wa_lo = (__bf16*)carve(WMAT * 2);
  __bf16* Uc_hi = (__bf16*)carve(WMAT * 2);
  __bf16* Uc_lo = (__bf16*)carve(WMAT * 2);
  __bf16* Wc_hi = (__bf16*)carve(WMAT * 2);
  __bf16* Wc_lo = (__bf16*)carve(WMAT * 2);
  __bf16* U_hi  = (__bf16*)carve(WMAT * 2);
  __bf16* U_lo  = (__bf16*)carve(WMAT * 2);
  __bf16* fc_hi = (__bf16*)carve((size_t)DOUT * HID * 2);
  __bf16* fc_lo = (__bf16*)carve((size_t)DOUT * HID * 2);

  __bf16* xA_hi = (__bf16*)carve((size_t)BT * DIN * 2);
  __bf16* xA_lo = (__bf16*)carve((size_t)BT * DIN * 2);
  __bf16* xB_hi = (__bf16*)carve((size_t)BT * DIN * 2);
  __bf16* xB_lo = (__bf16*)carve((size_t)BT * DIN * 2);

  float* pa = (float*)carve((size_t)BT * HID * 4);
  float* pc = (float*)carve((size_t)BT * HID * 4);
  float* pu = (float*)carve((size_t)BT * HID * 4);

  float*  hbuf[2];
  __bf16* hhib[2];
  __bf16* hlob[2];
  for (int i = 0; i < 2; ++i) {
    hbuf[i] = (float*)carve((size_t)BB * HID * 4);
    hhib[i] = (__bf16*)carve((size_t)BB * HID * 2);
    hlob[i] = (__bf16*)carve((size_t)BB * HID * 2);
  }

  const int thr = 256;
  auto conv = [&](const float* src, __bf16* hi, __bf16* lo, size_t n) {
    split_bf16_kernel<<<(unsigned)((n + thr - 1) / thr), thr, 0, stream>>>(src, hi, lo,
                                                                           (int)n);
  };
  conv(x,    xA_hi, xA_lo, (size_t)BT * DIN);
  conv(Ua_w, Ua_hi, Ua_lo, WMAT);
  conv(Wa_w, Wa_hi, Wa_lo, WMAT);
  conv(Uc_w, Uc_hi, Uc_lo, WMAT);
  conv(Wc_w, Wc_hi, Wc_lo, WMAT);
  conv(U_w,  U_hi,  U_lo,  WMAT);
  conv(fc_w, fc_hi, fc_lo, (size_t)DOUT * HID);

  __bf16 *curHi = xA_hi, *curLo = xA_lo, *nxtHi = xB_hi, *nxtLo = xB_lo;

  for (int l = 0; l < LL; ++l) {
    const size_t wo = (size_t)l * HID * DIN;
    const size_t bo = (size_t)l * HID;

    // batched input projections: pa/pc/pu = x@{Ua,Uc,U}^T + bias
    proj3_wmma_kernel<<<dim3(BT / 16, 2), 512, 0, stream>>>(
        curHi, curLo,
        Ua_hi + wo, Ua_lo + wo, Uc_hi + wo, Uc_lo + wo, U_hi + wo, U_lo + wo,
        Ua_b + bo, Uc_b + bo, U_b + bo,
        pa, pc, pu);

    zero_state_kernel<<<(BB * HID + 255) / 256, 256, 0, stream>>>(hbuf[0], hhib[0],
                                                                  hlob[0], BB * HID);

    for (int t = 0; t < TT; ++t) {
      const int cur = t & 1, nxt = (t + 1) & 1;
      nbrc_step_kernel<<<8, 512, 0, stream>>>(
          pa, pc, pu,
          Wa_hi + wo, Wa_lo + wo, Wc_hi + wo, Wc_lo + wo,
          Wa_b + bo, Wc_b + bo,
          hbuf[cur], hhib[cur], hlob[cur],
          hbuf[nxt], hhib[nxt], hlob[nxt],
          nxtHi, nxtLo, t);
    }

    // layer output becomes next layer input
    __bf16* t1 = curHi; curHi = nxtHi; nxtHi = t1;
    __bf16* t2 = curLo; curLo = nxtLo; nxtLo = t2;
  }

  // final FC over every timestep
  proj1_wmma_kernel<<<dim3(BT / 16, 2), 512, 0, stream>>>(
      curHi, curLo, fc_hi, fc_lo, fc_b, (float*)d_out);
}
